// HGNN_50345606644032
// MI455X (gfx1250) — matrix operand order repaired and checked by reading, not compiled, not run on hardware
//
#include <hip/hip_runtime.h>

// ---------------------------------------------------------------------------
// HGNN metapath aggregation for MI455X (gfx1250, wave32, WMMA bf16).
// Dominant cost: GRU gate GEMMs (~708 GFLOP) -> fused bf16 WMMA kernel with
// h kept in LDS (double buffered), W_hh streamed from L2.
// ---------------------------------------------------------------------------

#define N_NODES 20000
#define N_EDGES 100000
#define WALK_LEN 4
#define FEAT_DIM 256
#define HID 64
#define NUM_HEAD 8
#define H_RNN 512          // NUM_HEAD * HID
#define GATES 1536         // 3 * H_RNN
#define OUT_DIM 16

typedef __attribute__((ext_vector_type(16))) __bf16 v16bf;
typedef __attribute__((ext_vector_type(8)))  float  v8f;

union Frag {
  v16bf v;
  unsigned short u[16];
  uint4 q[2];
};

__device__ __forceinline__ unsigned short f2bf(float f) {
  unsigned int x = __float_as_uint(f);
  unsigned int r = 0x7FFFu + ((x >> 16) & 1u);   // round-to-nearest-even
  return (unsigned short)((x + r) >> 16);
}
__device__ __forceinline__ float bf2f(unsigned short h) {
  return __uint_as_float(((unsigned int)h) << 16);
}
__device__ __forceinline__ float sigm(float x) { return 1.0f / (1.0f + __expf(-x)); }
__device__ __forceinline__ float tanh_f(float x) { return 2.0f / (1.0f + __expf(-2.0f * x)) - 1.0f; }

// A operand: 16x32 bf16 tile, row-major source (ld in elements), k-offset kk.
// lane l: row = l&15; chunks [k0,k0+8) and [k0+16,k0+24), k0 = kk + (l>>4)*8.
__device__ __forceinline__ v16bf load_a16(const unsigned short* base, int ld, int kk, int lane) {
  const unsigned short* p = base + (lane & 15) * ld + kk + ((lane >> 4) << 3);
  Frag f;
  f.q[0] = *(const uint4*)(p);
  f.q[1] = *(const uint4*)(p + 16);
  return f.v;
}

// Same as load_a16 but converting from f32 source on the fly.
__device__ __forceinline__ v16bf load_a_f32(const float* base, int ld, int kk, int lane) {
  const float* p = base + (lane & 15) * ld + kk + ((lane >> 4) << 3);
  Frag f;
#pragma unroll
  for (int i = 0; i < 8; ++i) f.u[i] = f2bf(p[i]);
#pragma unroll
  for (int i = 0; i < 8; ++i) f.u[8 + i] = f2bf(p[16 + i]);
  return f.v;
}

// B operand: B[k,n] = W[n,k], W row-major (N,K) with row stride ld.
// lane l: col n = n0 + (l&15); chunks [k0,k0+8) and [k0+16,k0+24), k0 = kk + (l>>4)*8.
__device__ __forceinline__ v16bf load_b16(const unsigned short* w, int ld, int n0, int kk, int lane) {
  const unsigned short* p = w + (size_t)(n0 + (lane & 15)) * ld + kk + ((lane >> 4) << 3);
  Frag f;
  f.q[0] = *(const uint4*)(p);
  f.q[1] = *(const uint4*)(p + 16);
  return f.v;
}

__device__ __forceinline__ v8f wmma_bf16(v16bf a, v16bf b, v8f c) {
  return __builtin_amdgcn_wmma_f32_16x16x32_bf16(false, a, false, b, (short)0, c, false, false);
}

// --------------------------- elementwise kernels ---------------------------

__global__ void cvt_f32_bf16_kernel(const float* __restrict__ src,
                                    unsigned short* __restrict__ dst, int n) {
  int stride = gridDim.x * blockDim.x;
  for (int i = blockIdx.x * blockDim.x + threadIdx.x; i < n; i += stride)
    dst[i] = f2bf(src[i]);
}

__global__ void init_kernel(float* __restrict__ m, float* __restrict__ ssum,
                            float* __restrict__ ft) {
  int stride = gridDim.x * blockDim.x;
  int i0 = blockIdx.x * blockDim.x + threadIdx.x;
  for (int i = i0; i < N_NODES * NUM_HEAD; i += stride) { m[i] = -3.0e38f; ssum[i] = 0.0f; }
  for (int i = i0; i < N_NODES * H_RNN; i += stride) ft[i] = 0.0f;
}

// --------------------------- emb = x @ W_mlp.T + b -------------------------
// block: 64 threads (2 waves); each block: 16 nodes; each wave: 2 N-tiles.
__global__ void emb_kernel(const unsigned short* __restrict__ xbf,
                           const unsigned short* __restrict__ Wmlp,
                           const float* __restrict__ b_mlp,
                           unsigned short* __restrict__ embG) {
  int lane = threadIdx.x & 31;
  int wid = threadIdx.x >> 5;
  int m0 = blockIdx.x * 16;
  const unsigned short* A = xbf + (size_t)m0 * FEAT_DIM;
#pragma unroll
  for (int s = 0; s < 2; ++s) {
    int n0 = (wid * 2 + s) * 16;
    v8f acc = {};
#pragma unroll
    for (int kk = 0; kk < FEAT_DIM; kk += 32) {
      v16bf a = load_a16(A, FEAT_DIM, kk, lane);
      v16bf b = load_b16(Wmlp, FEAT_DIM, n0, kk, lane);
      acc = wmma_bf16(a, b, acc);
    }
    int c = n0 + (lane & 15);
    int g = lane >> 4;
    float bb = b_mlp[c];
#pragma unroll
    for (int r = 0; r < 8; ++r)
      embG[(size_t)(m0 + r + 8 * g) * HID + c] = f2bf(acc[r] + bb);
  }
}

// ------------------------------ fused GRU ----------------------------------
// block: 256 threads (8 waves), 16 edges. h (16x512 bf16) double-buffered in
// LDS. Each wave owns 4 hidden tiles (4*16 = 64 cols of H_RNN); per step it
// runs r/z/n gate WMMAs over K=64 (x_t) and K=512 (h) and applies the GRU
// update in registers. Attention scores fused at the end.
__global__ void __launch_bounds__(256) gru_kernel(
    const unsigned short* __restrict__ embG,
    const int* __restrict__ idx,
    const unsigned short* __restrict__ Wih,
    const unsigned short* __restrict__ Whh,
    const float* __restrict__ b_ih,
    const float* __restrict__ b_hh,
    const float* __restrict__ attn,
    unsigned short* __restrict__ hT,
    float* __restrict__ aG) {
  __shared__ unsigned short sh_h[2][16 * H_RNN];  // 2 * 16KB
  __shared__ unsigned short sh_x[WALK_LEN][16 * HID];  // 8KB

  int tid = threadIdx.x;
  int lane = tid & 31;
  int wid = tid >> 5;
  int e0 = blockIdx.x * 16;

  // gather x_t tiles: 64 (edge,t) pairs, 4 threads each copy 16 bf16.
  {
    int pair = tid >> 2;          // 0..63
    int el = pair >> 2;           // edge in tile
    int t = pair & 3;             // walk step
    int chunk = (tid & 3) * 16;   // 16 contiguous bf16
    int node = idx[(size_t)(e0 + el) * WALK_LEN + t];
    const uint4* src = (const uint4*)(embG + (size_t)node * HID + chunk);
    uint4* dst = (uint4*)(&sh_x[t][el * HID + chunk]);
    dst[0] = src[0];
    dst[1] = src[1];
  }
  // zero h0
  {
    uint4 z = {0, 0, 0, 0};
    uint4* h0q = (uint4*)(&sh_h[0][0]);
    for (int i = tid; i < 16 * H_RNN / 8; i += 256) h0q[i] = z;
  }
  __syncthreads();

  for (int t = 0; t < WALK_LEN; ++t) {
    const unsigned short* hcur = sh_h[t & 1];
    unsigned short* hnext = sh_h[(t + 1) & 1];
#pragma unroll
    for (int q = 0; q < 4; ++q) {
      int n0 = (wid * 4 + q) * 16;  // hidden tile column base
      v8f ar = {}, az = {}, axn = {}, ahn = {};
      // x_t @ W_ih.T  (K = 64)
#pragma unroll
      for (int kk = 0; kk < HID; kk += 32) {
        v16bf a = load_a16(&sh_x[t][0], HID, kk, lane);
        v16bf br = load_b16(Wih, HID, 0 * H_RNN + n0, kk, lane);
        v16bf bz = load_b16(Wih, HID, 1 * H_RNN + n0, kk, lane);
        v16bf bn = load_b16(Wih, HID, 2 * H_RNN + n0, kk, lane);
        ar = wmma_bf16(a, br, ar);
        az = wmma_bf16(a, bz, az);
        axn = wmma_bf16(a, bn, axn);
      }
      // h @ W_hh.T  (K = 512)
#pragma unroll 4
      for (int kk = 0; kk < H_RNN; kk += 32) {
        v16bf a = load_a16(hcur, H_RNN, kk, lane);
        v16bf br = load_b16(Whh, H_RNN, 0 * H_RNN + n0, kk, lane);
        v16bf bz = load_b16(Whh, H_RNN, 1 * H_RNN + n0, kk, lane);
        v16bf bn = load_b16(Whh, H_RNN, 2 * H_RNN + n0, kk, lane);
        ar = wmma_bf16(a, br, ar);
        az = wmma_bf16(a, bz, az);
        ahn = wmma_bf16(a, bn, ahn);
      }
      int c = n0 + (lane & 15);
      int g = lane >> 4;
      float br_ = b_ih[c] + b_hh[c];
      float bz_ = b_ih[H_RNN + c] + b_hh[H_RNN + c];
      float bxn_ = b_ih[2 * H_RNN + c];
      float bhn_ = b_hh[2 * H_RNN + c];
#pragma unroll
      for (int r = 0; r < 8; ++r) {
        int m = r + 8 * g;
        float rg = sigm(ar[r] + br_);
        float zg = sigm(az[r] + bz_);
        float ng = tanh_f(axn[r] + bxn_ + rg * (ahn[r] + bhn_));
        float ho = bf2f(hcur[m * H_RNN + c]);
        hnext[m * H_RNN + c] = f2bf((1.0f - zg) * ng + zg * ho);
      }
    }
    __syncthreads();
  }

  // final h is in sh_h[0] (WALK_LEN == 4). Write hT (bf16) + attention scores.
  const unsigned short* h0 = sh_h[0];
  {
    const uint4* src = (const uint4*)h0;
    uint4* dst = (uint4*)(hT + (size_t)e0 * H_RNN);
    for (int i = tid; i < 16 * H_RNN / 8; i += 256) dst[i] = src[i];
  }
  if (tid < 16 * NUM_HEAD) {
    int el = tid >> 3;
    int hd = tid & 7;
    float s = 0.0f;
#pragma unroll 8
    for (int j = 0; j < HID; ++j)
      s += bf2f(h0[el * H_RNN + hd * HID + j]) * attn[hd * HID + j];
    aG[(size_t)(e0 + el) * NUM_HEAD + hd] = (s > 0.0f) ? s : 0.01f * s;
  }
}

// ------------------------- segment softmax + scatter -----------------------

__device__ __forceinline__ void atomicMaxF(float* addr, float v) {
  if (v >= 0.0f)
    atomicMax((int*)addr, __float_as_int(v));
  else
    atomicMin((unsigned int*)addr, (unsigned int)__float_as_int(v));
}

__global__ void amax_kernel(const float* __restrict__ aG, const int* __restrict__ idx,
                            float* __restrict__ mG) {
  int i = blockIdx.x * blockDim.x + threadIdx.x;
  if (i >= N_EDGES * NUM_HEAD) return;
  int e = i >> 3, h = i & 7;
  int dst = idx[(size_t)e * WALK_LEN + (WALK_LEN - 1)];
  atomicMaxF(&mG[dst * NUM_HEAD + h], aG[i]);
}

__global__ void esum_kernel(float* __restrict__ aG, const int* __restrict__ idx,
                            const float* __restrict__ mG, float* __restrict__ ssum) {
  int i = blockIdx.x * blockDim.x + threadIdx.x;
  if (i >= N_EDGES * NUM_HEAD) return;
  int e = i >> 3, h = i & 7;
  int dst = idx[(size_t)e * WALK_LEN + (WALK_LEN - 1)];
  float ea = __expf(aG[i] - mG[dst * NUM_HEAD + h]);
  aG[i] = ea;  // reuse buffer for exp(a - m)
  atomicAdd(&ssum[dst * NUM_HEAD + h], ea);
}

__global__ void scatter_kernel(const unsigned short* __restrict__ hT,
                               const float* __restrict__ eaG,
                               const float* __restrict__ ssum,
                               const int* __restrict__ idx,
                               float* __restrict__ ft) {
  size_t i = (size_t)blockIdx.x * blockDim.x + threadIdx.x;
  if (i >= (size_t)N_EDGES * H_RNN) return;
  int e = (int)(i >> 9);
  int j = (int)(i & (H_RNN - 1));
  int h = j >> 6;
  int dst = idx[(size_t)e * WALK_LEN + (WALK_LEN - 1)];
  float alpha = eaG[(size_t)e * NUM_HEAD + h] / ssum[dst * NUM_HEAD + h];
  atomicAdd(&ft[(size_t)dst * H_RNN + j], bf2f(hT[i]) * alpha);
}

// ------------------- out = (ft @ W_emb.T + b) @ W_last.T + b ---------------
// block: 32 threads (1 wave), 16 nodes. rnn_emb kept in LDS as bf16.
__global__ void out_kernel(const float* __restrict__ ft,
                           const unsigned short* __restrict__ Wemb,
                           const float* __restrict__ b_emb,
                           const unsigned short* __restrict__ Wlast,
                           const float* __restrict__ b_last,
                           float* __restrict__ out) {
  __shared__ unsigned short sre[16 * HID];
  int lane = threadIdx.x;
  int m0 = blockIdx.x * 16;
  const float* A = ft + (size_t)m0 * H_RNN;
#pragma unroll
  for (int nt = 0; nt < 4; ++nt) {
    int n0 = nt * 16;
    v8f acc = {};
#pragma unroll 4
    for (int kk = 0; kk < H_RNN; kk += 32) {
      v16bf a = load_a_f32(A, H_RNN, kk, lane);
      v16bf b = load_b16(Wemb, H_RNN, n0, kk, lane);
      acc = wmma_bf16(a, b, acc);
    }
    int c = n0 + (lane & 15);
    int g = lane >> 4;
    float bb = b_emb[c];
#pragma unroll
    for (int r = 0; r < 8; ++r)
      sre[(r + 8 * g) * HID + c] = f2bf(acc[r] + bb);
  }
  __syncthreads();
  v8f acc = {};
#pragma unroll
  for (int kk = 0; kk < HID; kk += 32) {
    v16bf a = load_a16(sre, HID, kk, lane);
    v16bf b = load_b16(Wlast, HID, 0, kk, lane);
    acc = wmma_bf16(a, b, acc);
  }
  int c = lane & 15;
  int g = lane >> 4;
  float bb = b_last[c];
#pragma unroll
  for (int r = 0; r < 8; ++r)
    out[(size_t)(m0 + r + 8 * g) * OUT_DIM + c] = acc[r] + bb;
}

// ------------------------------- launcher ----------------------------------

extern "C" void kernel_launch(void* const* d_in, const int* in_sizes, int n_in,
                              void* d_out, int out_size, void* d_ws, size_t ws_size,
                              hipStream_t stream) {
  (void)in_sizes; (void)n_in; (void)out_size; (void)ws_size;
  const float* x      = (const float*)d_in[0];
  const float* W_mlp  = (const float*)d_in[1];
  const float* b_mlp  = (const float*)d_in[2];
  const float* W_ih   = (const float*)d_in[3];
  const float* W_hh   = (const float*)d_in[4];
  const float* b_ih   = (const float*)d_in[5];
  const float* b_hh   = (const float*)d_in[6];
  const float* attn   = (const float*)d_in[7];
  const float* W_emb  = (const float*)d_in[8];
  const float* b_emb  = (const float*)d_in[9];
  const float* W_last = (const float*)d_in[10];
  const float* b_last = (const float*)d_in[11];
  const int*   idx    = (const int*)d_in[12];
  float* out = (float*)d_out;

  // workspace carve-up (~163 MB total)
  char* ws = (char*)d_ws;
  size_t off = 0;
  auto carve = [&](size_t bytes) -> void* {
    off = (off + 255) & ~(size_t)255;
    void* p = ws + off;
    off += bytes;
    return p;
  };
  unsigned short* x_bf    = (unsigned short*)carve((size_t)N_NODES * FEAT_DIM * 2);
  unsigned short* Wmlp_bf = (unsigned short*)carve((size_t)HID * FEAT_DIM * 2);
  unsigned short* Wih_bf  = (unsigned short*)carve((size_t)GATES * HID * 2);
  unsigned short* Whh_bf  = (unsigned short*)carve((size_t)GATES * H_RNN * 2);
  unsigned short* Wemb_bf = (unsigned short*)carve((size_t)HID * H_RNN * 2);
  unsigned short* Wlast_bf= (unsigned short*)carve((size_t)OUT_DIM * HID * 2);
  unsigned short* emb_bf  = (unsigned short*)carve((size_t)N_NODES * HID * 2);
  unsigned short* hT_bf   = (unsigned short*)carve((size_t)N_EDGES * H_RNN * 2);
  float* aG   = (float*)carve((size_t)N_EDGES * NUM_HEAD * 4);
  float* mG   = (float*)carve((size_t)N_NODES * NUM_HEAD * 4);
  float* ssum = (float*)carve((size_t)N_NODES * NUM_HEAD * 4);
  float* ftG  = (float*)carve((size_t)N_NODES * H_RNN * 4);

  // 1) f32 -> bf16 conversions
  cvt_f32_bf16_kernel<<<2048, 256, 0, stream>>>(x, x_bf, N_NODES * FEAT_DIM);
  cvt_f32_bf16_kernel<<<64,   256, 0, stream>>>(W_mlp, Wmlp_bf, HID * FEAT_DIM);
  cvt_f32_bf16_kernel<<<384,  256, 0, stream>>>(W_ih, Wih_bf, GATES * HID);
  cvt_f32_bf16_kernel<<<2048, 256, 0, stream>>>(W_hh, Whh_bf, GATES * H_RNN);
  cvt_f32_bf16_kernel<<<128,  256, 0, stream>>>(W_emb, Wemb_bf, HID * H_RNN);
  cvt_f32_bf16_kernel<<<4,    256, 0, stream>>>(W_last, Wlast_bf, OUT_DIM * HID);

  // 2) init segment-max / segment-sum / ft accumulators
  init_kernel<<<2048, 256, 0, stream>>>(mG, ssum, ftG);

  // 3) node embedding MLP (WMMA)
  emb_kernel<<<N_NODES / 16, 64, 0, stream>>>(x_bf, Wmlp_bf, b_mlp, emb_bf);

  // 4) fused GRU over metapaths (WMMA) + attention scores
  gru_kernel<<<N_EDGES / 16, 256, 0, stream>>>(emb_bf, idx, Wih_bf, Whh_bf,
                                               b_ih, b_hh, attn, hT_bf, aG);

  // 5-7) segment softmax + weighted scatter
  amax_kernel<<<(N_EDGES * NUM_HEAD + 255) / 256, 256, 0, stream>>>(aG, idx, mG);
  esum_kernel<<<(N_EDGES * NUM_HEAD + 255) / 256, 256, 0, stream>>>(aG, idx, mG, ssum);
  scatter_kernel<<<(int)(((size_t)N_EDGES * H_RNN + 255) / 256), 256, 0, stream>>>(
      hT_bf, aG, ssum, idx, ftG);

  // 8) output MLP (WMMA)
  out_kernel<<<N_NODES / 16, 32, 0, stream>>>(ftG, Wemb_bf, b_emb, Wlast_bf, b_last, out);
}
